// DeepNeuralDecisionForest_40346922779036
// MI455X (gfx1250) — matrix operand order repaired and checked by reading, not compiled, not run on hardware
//
#include <hip/hip_runtime.h>
#include <hip/hip_bf16.h>
#include <math.h>

typedef __attribute__((ext_vector_type(16))) _Float16 v16h;
typedef __attribute__((ext_vector_type(8)))  _Float16 v8h;
typedef __attribute__((ext_vector_type(8)))  float    v8f;

#define NB      4096
#define NTREE   32
#define NLEAF   128
#define NLABEL  10
#define NFEAT   320
#define NH      50     // hidden dim (valid)
#define NHP     64     // hidden dim padded for WMMA K/N

// ---------------------------------------------------------------------------
// conv1: x(4096,1,28,28) -> relu(conv5x5 -> 10ch) -> maxpool2 -> h1(4096,10,12,12)
// ---------------------------------------------------------------------------
__global__ void k_conv1(const float* __restrict__ x, const float* __restrict__ w,
                        const float* __restrict__ bias, float* __restrict__ h1) {
  __shared__ float ws[250];
  __shared__ float bs[10];
  int tid = threadIdx.x;
  if (tid < 250) ws[tid] = w[tid];
  if (tid < 10)  bs[tid] = bias[tid];
  __syncthreads();
  int id = blockIdx.x * 256 + tid;
  const int total = NB * 10 * 12 * 12;
  if (id >= total) return;
  int px = id % 12, py = (id / 12) % 12, ch = (id / 144) % 10, b = id / 1440;
  const float* xb = x + b * 784;
  const float* wc = ws + ch * 25;
  float best = -1e30f;
#pragma unroll
  for (int dy = 0; dy < 2; ++dy)
#pragma unroll
    for (int dx = 0; dx < 2; ++dx) {
      float s = bs[ch];
      int iy0 = 2 * py + dy, ix0 = 2 * px + dx;
#pragma unroll
      for (int ky = 0; ky < 5; ++ky)
#pragma unroll
        for (int kx = 0; kx < 5; ++kx)
          s = fmaf(xb[(iy0 + ky) * 28 + ix0 + kx], wc[ky * 5 + kx], s);
      s = fmaxf(s, 0.0f);
      best = fmaxf(best, s);
    }
  h1[id] = best;
}

// ---------------------------------------------------------------------------
// conv2: h1(4096,10,12,12) -> relu(conv5x5 -> 20ch) -> maxpool2 -> feat f16 (4096,320)
// feat index matches reference reshape: f = ch*16 + py*4 + px
// ---------------------------------------------------------------------------
__global__ void k_conv2(const float* __restrict__ h1, const float* __restrict__ w,
                        const float* __restrict__ bias, _Float16* __restrict__ feat) {
  __shared__ float ws[5000];
  __shared__ float bs[20];
  int tid = threadIdx.x;
  for (int i = tid; i < 5000; i += 256) ws[i] = w[i];
  if (tid < 20) bs[tid] = bias[tid];
  __syncthreads();
  int id = blockIdx.x * 256 + tid;
  const int total = NB * 20 * 16;
  if (id >= total) return;
  int px = id & 3, py = (id >> 2) & 3, ch = (id >> 4) % 20, b = id / 320;
  const float* hb = h1 + b * 1440;
  const float* wc = ws + ch * 250;
  float best = -1e30f;
  for (int dy = 0; dy < 2; ++dy)
    for (int dx = 0; dx < 2; ++dx) {
      float s = bs[ch];
      int iy0 = 2 * py + dy, ix0 = 2 * px + dx;
      for (int ic = 0; ic < 10; ++ic) {
        const float* hi = hb + ic * 144;
        const float* wi = wc + ic * 25;
#pragma unroll
        for (int ky = 0; ky < 5; ++ky)
#pragma unroll
          for (int kx = 0; kx < 5; ++kx)
            s = fmaf(hi[(iy0 + ky) * 12 + ix0 + kx], wi[ky * 5 + kx], s);
      }
      s = fmaxf(s, 0.0f);
      best = fmaxf(best, s);
    }
  feat[b * NFEAT + ch * 16 + py * 4 + px] = (_Float16)best;
}

// ---------------------------------------------------------------------------
// Weight pre-transpose to f16, N-major [t][n][k] (zero padded), so that the
// WMMA B-operand per-lane fragment is a single contiguous 32B load.
// ---------------------------------------------------------------------------
__global__ void k_cvtw1(const float* __restrict__ w1, _Float16* __restrict__ w1t) {
  int id = blockIdx.x * 256 + threadIdx.x;
  const int total = NTREE * NHP * NFEAT;
  if (id >= total) return;
  int k = id % NFEAT;
  int n = (id / NFEAT) % NHP;
  int t = id / (NFEAT * NHP);
  float v = (n < NH) ? w1[(t * NFEAT + k) * NH + n] : 0.0f;
  w1t[id] = (_Float16)v;
}

__global__ void k_cvtw2(const float* __restrict__ w2, _Float16* __restrict__ w2t) {
  int id = blockIdx.x * 256 + threadIdx.x;
  const int total = NTREE * NLEAF * NHP;
  if (id >= total) return;
  int k = id % NHP;
  int n = (id / NHP) % NLEAF;
  int t = id / (NHP * NLEAF);
  float v = (k < NH) ? w2[(t * NH + k) * NLEAF + n] : 0.0f;
  w2t[id] = (_Float16)v;
}

// leaf_p = softmax(pi, axis=-1); one thread per (t,leaf) row of 10
__global__ void k_softmax_pi(const float* __restrict__ pi, float* __restrict__ leafp) {
  int id = blockIdx.x * 256 + threadIdx.x;
  if (id >= NTREE * NLEAF) return;
  const float* p = pi + id * NLABEL;
  float m = p[0];
#pragma unroll
  for (int c = 1; c < NLABEL; ++c) m = fmaxf(m, p[c]);
  float e[NLABEL];
  float s = 0.0f;
#pragma unroll
  for (int c = 0; c < NLABEL; ++c) { e[c] = __expf(p[c] - m); s += e[c]; }
  float inv = 1.0f / s;
#pragma unroll
  for (int c = 0; c < NLABEL; ++c) leafp[id * NLABEL + c] = e[c] * inv;
}

// ---------------------------------------------------------------------------
// FC1 (WMMA): th[t][b][h] = relu(feat[b,:] @ w1[t,:,:] + b1[t,:]) stored f16,
// h padded to 64 (pad stored as exact 0). One wave = one 16(M=b) x 16(N=h)
// tile; 4 waves/block cover h=64; K-loop over 320 in steps of 32.
// ---------------------------------------------------------------------------
__global__ void k_fc1(const _Float16* __restrict__ feat, const _Float16* __restrict__ w1t,
                      const float* __restrict__ b1, _Float16* __restrict__ th) {
  const int lane = threadIdx.x & 31;
  const int wave = threadIdx.x >> 5;    // 0..3 -> N tile
  const int b0   = blockIdx.x * 16;
  const int t    = blockIdx.y;
  const int n0   = wave * 16;
  const int half = lane >> 4;           // 0 or 1 (lane group)
  const int l16  = lane & 15;

  v8f acc = {};
  // A layout (16-bit A 16x32): lane holds row m=l16; k = k0 + 16*half + {0..15}
  const _Float16* arow = feat + (b0 + l16) * NFEAT + half * 16;
  // B layout (16-bit B 32x16): lane holds col n=l16; k = k0 + 16*half + {0..15}
  const _Float16* brow = w1t + (t * NHP + n0 + l16) * NFEAT + half * 16;
  for (int k0 = 0; k0 < NFEAT; k0 += 32) {
    v8h alo = *(const v8h*)(arow + k0);
    v8h ahi = *(const v8h*)(arow + k0 + 8);
    v16h A;
#pragma unroll
    for (int i = 0; i < 8; ++i) { A[i] = alo[i]; A[i + 8] = ahi[i]; }
    v16h Bv = *(const v16h*)(brow + k0);
    acc = __builtin_amdgcn_wmma_f32_16x16x32_f16(false, A, false, Bv,
                                                 (short)0, acc, false, false);
  }
#pragma unroll
  for (int r = 0; r < 8; ++r) {
    int m = r + 8 * half;          // C/D layout: vgpr r -> M=r (lanes 0-15) / r+8
    int n = n0 + l16;
    float v = acc[r];
    if (n < NH) v += b1[t * NH + n];
    v = fmaxf(v, 0.0f);
    if (n >= NH) v = 0.0f;         // exact zero padding for FC2's K dim
    th[(t * NB + b0 + m) * NHP + n] = (_Float16)v;
  }
}

// ---------------------------------------------------------------------------
// FC2 (WMMA) fused with sigmoid -> tree routing products -> leaf posterior
// reduction. One block = one (16-batch, tree) tile; 8 waves cover all 128
// leaves. dec never touches global memory (would be 64 MB otherwise).
// Writes deterministic per-tree partial[t][b][c].
// ---------------------------------------------------------------------------
__global__ void k_fc2(const _Float16* __restrict__ th, const _Float16* __restrict__ w2t,
                      const float* __restrict__ b2, const float* __restrict__ leafp,
                      float* __restrict__ partial) {
  __shared__ float dec_s[16][NLEAF + 1];   // +1 pad: stride 129 avoids bank conflicts
  __shared__ float mu_s[16][NLEAF + 1];
  __shared__ float lp_s[NLEAF * NLABEL];
  const int tid  = threadIdx.x;
  const int lane = tid & 31;
  const int wave = tid >> 5;              // 0..7 -> N tile (16 leaves each)
  const int b0   = blockIdx.x * 16;
  const int t    = blockIdx.y;
  const int n0   = wave * 16;
  const int half = lane >> 4;
  const int l16  = lane & 15;

  for (int i = tid; i < NLEAF * NLABEL; i += 256)
    lp_s[i] = leafp[t * NLEAF * NLABEL + i];

  v8f acc = {};
  const _Float16* arow = th + (t * NB + b0 + l16) * NHP + half * 16;
  const _Float16* brow = w2t + (t * NLEAF + n0 + l16) * NHP + half * 16;
#pragma unroll
  for (int k0 = 0; k0 < NHP; k0 += 32) {
    v8h alo = *(const v8h*)(arow + k0);
    v8h ahi = *(const v8h*)(arow + k0 + 8);
    v16h A;
#pragma unroll
    for (int i = 0; i < 8; ++i) { A[i] = alo[i]; A[i + 8] = ahi[i]; }
    v16h Bv = *(const v16h*)(brow + k0);
    acc = __builtin_amdgcn_wmma_f32_16x16x32_f16(false, A, false, Bv,
                                                 (short)0, acc, false, false);
  }
#pragma unroll
  for (int r = 0; r < 8; ++r) {
    int m = r + 8 * half;
    int n = n0 + l16;
    float z = acc[r] + b2[t * NLEAF + n];
    dec_s[m][n] = 1.0f / (1.0f + __expf(-z));
  }
  __syncthreads();

  // mu[b][j] = prod_d ( bit(6-d,j)==0 ? dec[node(d,j)] : 1-dec[node(d,j)] )
  {
    int bl = tid & 15;
    int jb = (tid >> 4) * 8;
#pragma unroll
    for (int jj = 0; jj < 8; ++jj) {
      int j = jb + jj;
      float mu = 1.0f;
#pragma unroll
      for (int d = 0; d <= 6; ++d) {
        int node = (1 << d) - 1 + (j >> (7 - d));
        float g = dec_s[bl][node];
        mu *= (((j >> (6 - d)) & 1) == 0) ? g : (1.0f - g);
      }
      mu_s[bl][j] = mu;
    }
  }
  __syncthreads();

  // per-tree partial: fixed-order (deterministic) reduction over 128 leaves
  if (tid < 16 * NLABEL) {
    int bl = tid / NLABEL, c = tid % NLABEL;
    float s = 0.0f;
    for (int j = 0; j < NLEAF; ++j) s = fmaf(mu_s[bl][j], lp_s[j * NLABEL + c], s);
    partial[(t * NB + b0 + bl) * NLABEL + c] = s;
  }
}

// out[b][c] = log( sum_t partial / (NLEAF*NTREE) )
__global__ void k_finalize(const float* __restrict__ partial, float* __restrict__ out) {
  int id = blockIdx.x * 256 + threadIdx.x;
  if (id >= NB * NLABEL) return;
  int c = id % NLABEL, b = id / NLABEL;
  float s = 0.0f;
  for (int t = 0; t < NTREE; ++t) s += partial[(t * NB + b) * NLABEL + c];
  out[id] = __logf(s * (1.0f / (NLEAF * NTREE)));
}

// ---------------------------------------------------------------------------
// Workspace layout (bytes, 256-aligned), total ~50.2 MB
// ---------------------------------------------------------------------------
#define OFF_H1    0u          // 4096*10*144 f32      = 23,592,960
#define OFF_FEAT  23592960u   // 4096*320 f16         =  2,621,440
#define OFF_W1T   26214400u   // 32*64*320 f16        =  1,310,720
#define OFF_W2T   27525120u   // 32*128*64 f16        =    524,288
#define OFF_TH    28049408u   // 32*4096*64 f16       = 16,777,216
#define OFF_LEAFP 44826624u   // 32*128*10 f32        =    163,840
#define OFF_PART  44990464u   // 32*4096*10 f32       =  5,242,880

extern "C" void kernel_launch(void* const* d_in, const int* in_sizes, int n_in,
                              void* d_out, int out_size, void* d_ws, size_t ws_size,
                              hipStream_t stream) {
  (void)in_sizes; (void)n_in; (void)out_size; (void)ws_size;
  const float* x   = (const float*)d_in[0];
  const float* c1w = (const float*)d_in[1];
  const float* c1b = (const float*)d_in[2];
  const float* c2w = (const float*)d_in[3];
  const float* c2b = (const float*)d_in[4];
  const float* w1  = (const float*)d_in[5];
  const float* b1  = (const float*)d_in[6];
  const float* w2  = (const float*)d_in[7];
  const float* b2  = (const float*)d_in[8];
  const float* pi  = (const float*)d_in[9];
  float* out = (float*)d_out;

  char* ws = (char*)d_ws;
  float*    h1    = (float*)(ws + OFF_H1);
  _Float16* feat  = (_Float16*)(ws + OFF_FEAT);
  _Float16* w1t   = (_Float16*)(ws + OFF_W1T);
  _Float16* w2t   = (_Float16*)(ws + OFF_W2T);
  _Float16* th    = (_Float16*)(ws + OFF_TH);
  float*    leafp = (float*)(ws + OFF_LEAFP);
  float*    part  = (float*)(ws + OFF_PART);

  k_conv1<<<dim3((NB * 10 * 144 + 255) / 256), 256, 0, stream>>>(x, c1w, c1b, h1);
  k_cvtw1<<<dim3((NTREE * NHP * NFEAT + 255) / 256), 256, 0, stream>>>(w1, w1t);
  k_cvtw2<<<dim3((NTREE * NLEAF * NHP + 255) / 256), 256, 0, stream>>>(w2, w2t);
  k_softmax_pi<<<dim3((NTREE * NLEAF + 255) / 256), 256, 0, stream>>>(pi, leafp);
  k_conv2<<<dim3((NB * 20 * 16 + 255) / 256), 256, 0, stream>>>(h1, c2w, c2b, feat);
  k_fc1<<<dim3(NB / 16, NTREE), 128, 0, stream>>>(feat, w1t, b1, th);
  k_fc2<<<dim3(NB / 16, NTREE), 256, 0, stream>>>(th, w2t, b2, leafp, part);
  k_finalize<<<dim3((NB * NLABEL + 255) / 256), 256, 0, stream>>>(part, out);
}
